// L0ConjunctionLayer_1520418423390
// MI455X (gfx1250) — matrix operand order repaired and checked by reading, not compiled, not run on hardware
//
#include <hip/hip_runtime.h>

// L0 hard-concrete conjunction layer, WMMA-accelerated via log-series GEMM.
//
//   out[b,o] = prod_i (1 - (1 - x[b,i]*z[i]) * w[i,o]),  z = hardtanh(sigmoid(q)*1.2 - 0.1)
//
// With a = 1 - x*z in [0,1] and w in [0,0.1], u = a*w <= 0.1:
//   log out = sum_i log1p(-u) = -sum_{k=1..4} (1/k) sum_i a_i^k w_io^k   (err <= u^5/5)
// => one f16 GEMM [2048 x 4*512] x [4*512 x 256], f32 accumulate, exp(-S) epilogue.
//
// MI455X plan: power planes A^k precomputed to f16 (8MB ws) so the GEMM loop has no
// VALU power-chain competing with the XDL pipe. Per block, the shared 128x32 A slab
// is staged into LDS with global_load_async_to_lds_b128 (ASYNCcnt, double-buffered)
// -- 4 N-waves share it, cutting L2 traffic 4x and overlapping DMA with WMMA.

typedef __attribute__((ext_vector_type(16))) _Float16 v16h;
typedef __attribute__((ext_vector_type(8)))  _Float16 v8h;
typedef __attribute__((ext_vector_type(8)))  float    v8f;

typedef __attribute__((address_space(3))) unsigned char lds_uchar;

#define B_DIM 2048
#define I_DIM 512
#define O_DIM 256
#define NTERMS 4

#define LDS_ROW    80                    // 64B of slab row + 16B pad (bank spread)
#define SLAB_BYTES (128 * LDS_ROW)       // 10 KB per buffer

// ---------------------------------------------------------------------------
// Prep 1: a = 1 - x * clamp(sigmoid(q)*1.2 - 0.1, 0, 1); emit a, a^2, a^3, a^4
// as four f16 planes [t][B][I].
// ---------------------------------------------------------------------------
__global__ __launch_bounds__(256) void prep_a_kernel(const float* __restrict__ x,
                                                     const float* __restrict__ qz,
                                                     _Float16* __restrict__ a1) {
  int idx = blockIdx.x * blockDim.x + threadIdx.x;   // [0, B*I)
  int i = idx & (I_DIM - 1);
  float q  = qz[i];
  float pi = 1.0f / (1.0f + __expf(-q));
  float z  = pi * 1.2f - 0.1f;
  z = fminf(fmaxf(z, 0.0f), 1.0f);
  float a  = 1.0f - x[idx] * z;
  float a2 = a * a, a3 = a2 * a, a4 = a2 * a2;
  const size_t P = (size_t)B_DIM * I_DIM;
  a1[idx]         = (_Float16)a;
  a1[P + idx]     = (_Float16)a2;
  a1[2 * P + idx] = (_Float16)a3;
  a1[3 * P + idx] = (_Float16)a4;
}

// ---------------------------------------------------------------------------
// Prep 2: Wt[t][j][i] = w[i,j]^(t+1) / (t+1)  -> f16, TRANSPOSED so the WMMA
// B fragment (col-per-lane, contiguous K) is a contiguous 32B load per lane.
// ---------------------------------------------------------------------------
__global__ __launch_bounds__(256) void prep_w_kernel(const float* __restrict__ w,
                                                     _Float16* __restrict__ wt) {
  int idx = blockIdx.x * blockDim.x + threadIdx.x;   // [0, NTERMS*O*I)
  int t = idx >> 17;                 // I*O = 131072 = 1<<17
  int j = (idx >> 9) & (O_DIM - 1);
  int i = idx & (I_DIM - 1);
  float w1 = w[i * O_DIM + j];
  float v = w1;
  if (t >= 1) v *= w1;
  if (t >= 2) v *= w1;
  if (t >= 3) v *= w1;
  v *= (1.0f / (float)(t + 1));
  wt[idx] = (_Float16)v;
}

// ---------------------------------------------------------------------------
__device__ __forceinline__ void wait_async0() {
  asm volatile("s_wait_asynccnt 0x0" ::: "memory");
}

__device__ __forceinline__ void async_b128_to_lds(const _Float16* g, unsigned lds_off) {
  // GLOBAL_LOAD_ASYNC_TO_LDS_B128: vdst = VGPR with LDS byte address, GV addressing.
  asm volatile("global_load_async_to_lds_b128 %0, %1, off"
               :: "v"(lds_off), "v"(g)
               : "memory");
}

// ---------------------------------------------------------------------------
// GEMM + exp epilogue. 8 waves/block (2 along M x 4 along N); each wave owns a
// 64(M) x 16(N) tile: 4 M-subtiles reuse one B fragment per k-step (4 WMMAs per
// 32B B load). A slab (128 rows x 32 K, f16) double-buffered in LDS via async
// DMA. Grid: (2048/128, 256/64) = 16 x 4 blocks, 256 threads.
// ---------------------------------------------------------------------------
__global__ __launch_bounds__(256) void l0conj_wmma_kernel(const _Float16* __restrict__ A1,
                                                          const _Float16* __restrict__ Wt,
                                                          float* __restrict__ out) {
  __shared__ alignas(16) unsigned char lds_raw[2 * SLAB_BYTES];

  const int tid   = threadIdx.x;
  const int lane  = tid & 31;
  const int wave  = tid >> 5;
  const int r     = lane & 15;   // A row / B column / C column within tile
  const int h     = lane >> 4;   // lane half (K sub-block / C row half)
  const int waveM = wave & 1;    // 2 waves along M
  const int waveN = wave >> 1;   // 4 waves along N
  const int rowBlock = blockIdx.x * 128;
  const int col      = blockIdx.y * 64 + waveN * 16 + r;  // output column (row of Wt)

  // async-loader role: each thread moves 32B of the 128x64B slab
  const int ldRow  = tid >> 1;       // 0..127
  const int ldHalf = tid & 1;        // which 32B half of the 64B row

  const _Float16* brow = Wt + (size_t)col * I_DIM;   // term 0; terms step O*I
  const int myRow = waveM * 64 + r;                  // this lane's base row in slab

  v8f acc0 = {}, acc1 = {}, acc2 = {}, acc3 = {};

  auto issue_slab = [&](int kk, int buf) {
    const int t  = kk >> 4;            // term index 0..3
    const int ki = (kk & 15) << 5;     // K offset within term, 0..480
    const _Float16* src = A1 + (size_t)t * (B_DIM * I_DIM)
                        + (size_t)(rowBlock + ldRow) * I_DIM + ki + ldHalf * 16;
    const unsigned dst = (unsigned)(unsigned long long)
        (lds_uchar*)&lds_raw[buf * SLAB_BYTES + ldRow * LDS_ROW + ldHalf * 32];
    async_b128_to_lds(src,     dst);
    async_b128_to_lds(src + 8, dst + 16);
  };

  issue_slab(0, 0);
  wait_async0();
  __syncthreads();

  for (int kk = 0; kk < 4 * (I_DIM / 32); ++kk) {
    const int buf = kk & 1;
    if (kk + 1 < 4 * (I_DIM / 32)) issue_slab(kk + 1, buf ^ 1);  // overlap DMA w/ WMMA

    const int t  = kk >> 4;
    const int ki = (kk & 15) << 5;

    // B fragment: N = lane&15, K = ki + (lane>>4)*16 + e  -> contiguous 32B (L2-hot)
    v16h bf = *(const v16h*)(brow + (size_t)t * (O_DIM * I_DIM) + ki + h * 16);

    // A fragments from LDS slab: elems 0-15 -> K=h*16+e
    const unsigned char* sb = &lds_raw[buf * SLAB_BYTES];
    v8h al0 = *(const v8h*)(sb + (myRow +  0) * LDS_ROW + h * 32);
    v8h ah0 = *(const v8h*)(sb + (myRow +  0) * LDS_ROW + 16 + h * 32);
    v8h al1 = *(const v8h*)(sb + (myRow + 16) * LDS_ROW + h * 32);
    v8h ah1 = *(const v8h*)(sb + (myRow + 16) * LDS_ROW + 16 + h * 32);
    v8h al2 = *(const v8h*)(sb + (myRow + 32) * LDS_ROW + h * 32);
    v8h ah2 = *(const v8h*)(sb + (myRow + 32) * LDS_ROW + 16 + h * 32);
    v8h al3 = *(const v8h*)(sb + (myRow + 48) * LDS_ROW + h * 32);
    v8h ah3 = *(const v8h*)(sb + (myRow + 48) * LDS_ROW + 16 + h * 32);

    v16h af0 = __builtin_shufflevector(al0, ah0, 0,1,2,3,4,5,6,7,8,9,10,11,12,13,14,15);
    v16h af1 = __builtin_shufflevector(al1, ah1, 0,1,2,3,4,5,6,7,8,9,10,11,12,13,14,15);
    v16h af2 = __builtin_shufflevector(al2, ah2, 0,1,2,3,4,5,6,7,8,9,10,11,12,13,14,15);
    v16h af3 = __builtin_shufflevector(al3, ah3, 0,1,2,3,4,5,6,7,8,9,10,11,12,13,14,15);

    acc0 = __builtin_amdgcn_wmma_f32_16x16x32_f16(false, af0, false, bf, (short)0, acc0, false, false);
    acc1 = __builtin_amdgcn_wmma_f32_16x16x32_f16(false, af1, false, bf, (short)0, acc1, false, false);
    acc2 = __builtin_amdgcn_wmma_f32_16x16x32_f16(false, af2, false, bf, (short)0, acc2, false, false);
    acc3 = __builtin_amdgcn_wmma_f32_16x16x32_f16(false, af3, false, bf, (short)0, acc3, false, false);

    wait_async0();     // next slab landed (issued before compute -> overlapped)
    __syncthreads();   // all waves done reading buf before it is overwritten
  }

  // Epilogue: C layout VGPR v -> M = v + 8*(lane>>4), N = lane&15; out = exp(-S)
#pragma unroll
  for (int v = 0; v < 8; ++v) {
    int m0 = rowBlock + waveM * 64 + h * 8 + v;
    out[(size_t)(m0 +  0) * O_DIM + col] = __expf(-acc0[v]);
    out[(size_t)(m0 + 16) * O_DIM + col] = __expf(-acc1[v]);
    out[(size_t)(m0 + 32) * O_DIM + col] = __expf(-acc2[v]);
    out[(size_t)(m0 + 48) * O_DIM + col] = __expf(-acc3[v]);
  }
}

// ---------------------------------------------------------------------------
extern "C" void kernel_launch(void* const* d_in, const int* in_sizes, int n_in,
                              void* d_out, int out_size, void* d_ws, size_t ws_size,
                              hipStream_t stream) {
  const float* x  = (const float*)d_in[0];   // [2048, 512]
  const float* w  = (const float*)d_in[1];   // [512, 256]
  const float* qz = (const float*)d_in[2];   // [512]
  float* out = (float*)d_out;                // [2048, 256]

  // ws layout: Wt f16 [4][256][512] (1 MB) | A planes f16 [4][2048][512] (8 MB)
  _Float16* wt = (_Float16*)d_ws;
  _Float16* a1 = (_Float16*)((char*)d_ws +
                 (size_t)NTERMS * O_DIM * I_DIM * sizeof(_Float16));

  prep_a_kernel<<<(B_DIM * I_DIM) / 256, 256, 0, stream>>>(x, qz, a1);
  prep_w_kernel<<<(NTERMS * I_DIM * O_DIM) / 256, 256, 0, stream>>>(w, wt);

  dim3 grid(B_DIM / 128, O_DIM / 64);
  l0conj_wmma_kernel<<<grid, 256, 0, stream>>>(a1, wt, out);
}
